// NodeLevelAttentionLayer_27934467293579
// MI455X (gfx1250) — matrix operand order repaired and checked by reading, not compiled, not run on hardware
//
#include <hip/hip_runtime.h>
#include <hip/hip_bf16.h>
#include <math.h>

// ---------------------------------------------------------------------------
// GAT-style layer, fused one-pass formulation:
//   out_row = lrelu(nan_to_num( (sum_j w_ij * v_j) / (sum_j w_ij * sum_j Adj_ij) ))
//   with w_ij = exp(lrelu(rowscore_i + colscore_j)) * Adj_ij
// Heavy op done with v_wmma_f32_16x16x32_f16 (f16 A/B, f32 accum).
// ---------------------------------------------------------------------------

typedef _Float16 v16h __attribute__((ext_vector_type(16)));
typedef _Float16 v8h  __attribute__((ext_vector_type(8)));
typedef float    v8f  __attribute__((ext_vector_type(8)));
typedef float    v4f  __attribute__((ext_vector_type(4)));

#define NNODE 6144
#define NEDGE 12288
#define FN    128
#define FE    64
#define DNN   64
#define DEE   64

// ---------------- phase 1: small GEMMs + f16 transposes --------------------

__global__ __launch_bounds__(256) void k_h(const float* __restrict__ nf,
                                           const float* __restrict__ wn,
                                           float* __restrict__ h,
                                           _Float16* __restrict__ hT) {
  int idx = blockIdx.x * 256 + threadIdx.x;
  if (idx >= NNODE * DNN) return;
  int r = idx >> 6, c = idx & 63;
  const float* a = nf + (size_t)r * FN;
  float acc = 0.f;
#pragma unroll 8
  for (int k = 0; k < FN; ++k) acc += a[k] * wn[k * DNN + c];
  h[idx] = acc;
  hT[(size_t)c * NNODE + r] = (_Float16)acc;
}

__global__ __launch_bounds__(256) void k_g(const float* __restrict__ ef,
                                           const float* __restrict__ we,
                                           float* __restrict__ g,
                                           _Float16* __restrict__ gT) {
  int idx = blockIdx.x * 256 + threadIdx.x;
  if (idx >= NEDGE * DEE) return;
  int r = idx >> 6, c = idx & 63;
  const float* a = ef + (size_t)r * FE;
  float acc = 0.f;
#pragma unroll 8
  for (int k = 0; k < FE; ++k) acc += a[k] * we[k * DEE + c];
  g[idx] = acc;
  gT[(size_t)c * NEDGE + r] = (_Float16)acc;
}

__global__ __launch_bounds__(256) void k_scores_node(const float* __restrict__ h,
                                                     const float* __restrict__ pvn,
                                                     const float* __restrict__ pve,
                                                     float* __restrict__ f,
                                                     float* __restrict__ s,
                                                     float* __restrict__ fb) {
  int r = blockIdx.x * 256 + threadIdx.x;
  if (r >= NNODE) return;
  const float* hr = h + (size_t)r * DNN;
  float af = 0.f, as = 0.f, ab = 0.f;
#pragma unroll 8
  for (int c = 0; c < DNN; ++c) {
    float v = hr[c];
    af += v * pvn[c];
    as += v * pvn[DNN + c];
    ab += v * pve[c];
  }
  f[r] = af; s[r] = as; fb[r] = ab;
}

__global__ __launch_bounds__(256) void k_scores_edge(const float* __restrict__ g,
                                                     const float* __restrict__ pve,
                                                     float* __restrict__ gb) {
  int r = blockIdx.x * 256 + threadIdx.x;
  if (r >= NEDGE) return;
  const float* gr = g + (size_t)r * DEE;
  float ab = 0.f;
#pragma unroll 8
  for (int c = 0; c < DEE; ++c) ab += gr[c] * pve[DNN + c];
  gb[r] = ab;
}

// ---------------- phase 2: fused attention + WMMA aggregation --------------
// One block = 16 output rows. 8 waves split the j dimension (chunks of 32).
// Per chunk per wave: build 16x32 f16 W fragment in registers (exact CDNA5
// A-matrix layout), 4x v_wmma_f32_16x16x32_f16 against hT/gT columns.

__global__ __launch_bounds__(256) void k_attn(const float* __restrict__ Adj, int nj,
                                              const float* __restrict__ rowsc,
                                              const float* __restrict__ colsc,
                                              const _Float16* __restrict__ bT,
                                              float* __restrict__ out, int colOff) {
  __shared__ float Cs[16 * 64];
  __shared__ float dens[16];
  __shared__ float degs[16];

  const int tid = threadIdx.x;
  for (int i = tid; i < 16 * 64; i += 256) Cs[i] = 0.f;
  if (tid < 16) { dens[tid] = 0.f; degs[tid] = 0.f; }
  __syncthreads();

  const int wave = tid >> 5;
  const int lane = tid & 31;
  const int half = lane >> 4;   // 0: K 0..7 & 16..23 ; 1: K 8..15 & 24..31
  const int lm   = lane & 15;
  const int i0   = blockIdx.x << 4;
  const int row  = i0 + lm;
  const float fi = rowsc[row];

  v8f acc0 = {}, acc1 = {}, acc2 = {}, acc3 = {};
  float den = 0.f, deg = 0.f;

  const float* Arow = Adj + (size_t)row * nj;
  const int nchunk = nj >> 5;

  for (int c = wave; c < nchunk; c += 8) {
    const int j0 = c << 5;
    const int kb = j0 + (half << 3);

    // adjacency tile: streamed once, keep out of close caches (NT hint)
    const v4f a0 = __builtin_nontemporal_load((const v4f*)(Arow + kb));
    const v4f a1 = __builtin_nontemporal_load((const v4f*)(Arow + kb + 4));
    const v4f a2 = __builtin_nontemporal_load((const v4f*)(Arow + kb + 16));
    const v4f a3 = __builtin_nontemporal_load((const v4f*)(Arow + kb + 20));
    const v4f s0 = *(const v4f*)(colsc + kb);
    const v4f s1 = *(const v4f*)(colsc + kb + 4);
    const v4f s2 = *(const v4f*)(colsc + kb + 16);
    const v4f s3 = *(const v4f*)(colsc + kb + 20);

    const float av[16] = {a0[0], a0[1], a0[2], a0[3], a1[0], a1[1], a1[2], a1[3],
                          a2[0], a2[1], a2[2], a2[3], a3[0], a3[1], a3[2], a3[3]};
    const float sv[16] = {s0[0], s0[1], s0[2], s0[3], s1[0], s1[1], s1[2], s1[3],
                          s2[0], s2[1], s2[2], s2[3], s3[0], s3[1], s3[2], s3[3]};

    v16h af;
#pragma unroll
    for (int e = 0; e < 16; ++e) {
      float x = fi + sv[e];
      x = x > 0.f ? x : 0.01f * x;          // leaky_relu, slope 0.01
      float w = __expf(x) * av[e];          // v_exp_f32 (TRANS, co-exec w/ XDL)
      den += w;
      deg += av[e];
      af[e] = (_Float16)w;                  // CDNA5 16-bit A-matrix element order
    }

    // B fragments: column n = gi*16 + lm of bT, K = 8*half+0..7 and 16+8*half+0..7
    const _Float16* bbase = bT + (size_t)lm * nj + j0 + (half << 3);
    {
      const v8h* bp = (const v8h*)bbase;
      v16h bf = __builtin_shufflevector(bp[0], bp[2], 0,1,2,3,4,5,6,7,8,9,10,11,12,13,14,15);
      acc0 = __builtin_amdgcn_wmma_f32_16x16x32_f16(false, af, false, bf, (short)0, acc0, false, false);
    }
    {
      const v8h* bp = (const v8h*)(bbase + (size_t)16 * nj);
      v16h bf = __builtin_shufflevector(bp[0], bp[2], 0,1,2,3,4,5,6,7,8,9,10,11,12,13,14,15);
      acc1 = __builtin_amdgcn_wmma_f32_16x16x32_f16(false, af, false, bf, (short)0, acc1, false, false);
    }
    {
      const v8h* bp = (const v8h*)(bbase + (size_t)32 * nj);
      v16h bf = __builtin_shufflevector(bp[0], bp[2], 0,1,2,3,4,5,6,7,8,9,10,11,12,13,14,15);
      acc2 = __builtin_amdgcn_wmma_f32_16x16x32_f16(false, af, false, bf, (short)0, acc2, false, false);
    }
    {
      const v8h* bp = (const v8h*)(bbase + (size_t)48 * nj);
      v16h bf = __builtin_shufflevector(bp[0], bp[2], 0,1,2,3,4,5,6,7,8,9,10,11,12,13,14,15);
      acc3 = __builtin_amdgcn_wmma_f32_16x16x32_f16(false, af, false, bf, (short)0, acc3, false, false);
    }
  }

  // cross-wave reduction in LDS (ds_add_f32). Lanes L and L+16 share a row.
  atomicAdd(&dens[lm], den);
  atomicAdd(&degs[lm], deg);
#pragma unroll
  for (int v = 0; v < 8; ++v) {
    const int m = v + (half << 3);          // C/D layout: M = vgpr + 8*half
    atomicAdd(&Cs[m * 64 +  0 + lm], acc0[v]);
    atomicAdd(&Cs[m * 64 + 16 + lm], acc1[v]);
    atomicAdd(&Cs[m * 64 + 32 + lm], acc2[v]);
    atomicAdd(&Cs[m * 64 + 48 + lm], acc3[v]);
  }
  __syncthreads();

  // finalize: / (softmax-denominator * degree), nan_to_num, leaky_relu
  for (int i = tid; i < 16 * 64; i += 256) {
    const int m = i >> 6;
    float d = dens[m] * degs[m];
    float v = Cs[i] / d;
    if (__builtin_isnan(v)) v = 0.f;
    v = fminf(fmaxf(v, -3.4028234663852886e38f), 3.4028234663852886e38f);
    v = v > 0.f ? v : 0.01f * v;
    out[(size_t)(i0 + m) * (DNN + DEE) + colOff + (i & 63)] = v;
  }
}

// ---------------------------------------------------------------------------

extern "C" void kernel_launch(void* const* d_in, const int* in_sizes, int n_in,
                              void* d_out, int out_size, void* d_ws, size_t ws_size,
                              hipStream_t stream) {
  (void)in_sizes; (void)n_in; (void)out_size; (void)ws_size;

  const float* nf  = (const float*)d_in[0];  // (6144,128)
  const float* ef  = (const float*)d_in[1];  // (12288,64)
  const float* A   = (const float*)d_in[2];  // (6144,6144)
  const float* B   = (const float*)d_in[3];  // (6144,12288)
  const float* wn  = (const float*)d_in[4];  // (128,64)
  const float* we  = (const float*)d_in[5];  // (64,64)
  const float* pvn = (const float*)d_in[6];  // (128,)
  const float* pve = (const float*)d_in[7];  // (128,)
  float* out = (float*)d_out;                // (6144,128)

  // workspace carve-up (~6.9 MB), all offsets 256B-aligned
  char* ws = (char*)d_ws;
  float* h  = (float*)ws;                 ws += sizeof(float) * NNODE * DNN;   // 1.5 MB
  float* g  = (float*)ws;                 ws += sizeof(float) * NEDGE * DEE;   // 3.0 MB
  float* f  = (float*)ws;                 ws += sizeof(float) * NNODE;
  float* s  = (float*)ws;                 ws += sizeof(float) * NNODE;
  float* fb = (float*)ws;                 ws += sizeof(float) * NNODE;
  float* gb = (float*)ws;                 ws += sizeof(float) * NEDGE;
  _Float16* hT = (_Float16*)ws;           ws += sizeof(_Float16) * NNODE * DNN; // 0.77 MB
  _Float16* gT = (_Float16*)ws;           // 1.5 MB

  k_h<<<(NNODE * DNN + 255) / 256, 256, 0, stream>>>(nf, wn, h, hT);
  k_g<<<(NEDGE * DEE + 255) / 256, 256, 0, stream>>>(ef, we, g, gT);
  k_scores_node<<<(NNODE + 255) / 256, 256, 0, stream>>>(h, pvn, pve, f, s, fb);
  k_scores_edge<<<(NEDGE + 255) / 256, 256, 0, stream>>>(g, pve, gb);

  // node-node attention -> output cols [0,64)
  k_attn<<<NNODE / 16, 256, 0, stream>>>(A, NNODE, f, s, hT, out, 0);
  // node-edge attention -> output cols [64,128)
  k_attn<<<NNODE / 16, 256, 0, stream>>>(B, NEDGE, fb, gb, gT, out, DNN);
}
